// TrianDecoderUnroll_16956530884973
// MI455X (gfx1250) — matrix operand order repaired and checked by reading, not compiled, not run on hardware
//
#include <hip/hip_runtime.h>

#define BB 64
#define TT 64
#define HH 1024
#define VV 32000
#define EE 512

typedef __bf16 bf16;
typedef __attribute__((ext_vector_type(8)))  __bf16 v8bf;
typedef __attribute__((ext_vector_type(16))) __bf16 v16bf;
typedef __attribute__((ext_vector_type(8)))  float  v8f;

union AFrag { v16bf v; v8bf h[2]; };

// ---------------------------------------------------------------------------
// Pack a [K,N] row-major fp32 weight into bf16 WMMA B-fragments.
// B 32x16 16-bit layout (ISA 7.12.2): within a 32-K chunk, lanes 0-15 hold
// column N = lane with K = 0..7/16..23, lanes 16-31 hold column N = lane-16 with
// K = 8..15/24..31.  Packed element index = nt*(K*16) + kt*512 + lane*16 + e.
// ---------------------------------------------------------------------------
__global__ void pack_weights(const float* __restrict__ src, bf16* __restrict__ dst,
                             int K, int N)
{
    size_t tid = (size_t)blockIdx.x * blockDim.x + threadIdx.x;
    size_t total = (size_t)K * N;
    if (tid >= total) return;
    size_t ntSize = (size_t)K * 16;
    size_t nt  = tid / ntSize;
    size_t rem = tid - nt * ntSize;
    int kt   = (int)(rem >> 9);       // /512
    int q    = (int)(rem & 511);
    int lane = q >> 4;
    int e    = q & 15;
    int k = kt * 32 + ((lane >= 16) ? 8 : 0) + (e & 7) + ((e >> 3) << 4);
    int n = (int)nt * 16 + (lane & 15);
    dst[tid] = (bf16)src[(size_t)k * N + n];
}

// ---------------------------------------------------------------------------
// Software-pipelined WMMA panel: acc[0..3] += A[16 rows, K] @ B_packed[K, 64].
// Register double-buffering so next k-step's fragment loads are in flight
// while the current step's 4 WMMAs issue (breaks the s_wait_loadcnt 0 chain).
// Requires K % 64 == 0 (true: K is 512 or 1024).
// ---------------------------------------------------------------------------
__device__ __forceinline__ void gemm_panel(
    const bf16* __restrict__ A, int mA, int kb, int K,
    const bf16* __restrict__ Bt, v8f acc[4])
{
    const bf16* __restrict__ Arow = A + (size_t)mA * K + kb;
    const size_t ntStride = (size_t)K * 16;

    AFrag a0, a1;
    v16bf b0[4], b1[4];

    auto loadA = [&](AFrag& a, int kt) {
        a.h[0] = *(const v8bf*)(Arow + kt);
        a.h[1] = *(const v8bf*)(Arow + kt + 16);
    };
    auto loadB = [&](v16bf* b, int kt) {
        const size_t kOff = (size_t)(kt >> 5) << 9;     // *512
#pragma unroll
        for (int nn = 0; nn < 4; ++nn)
            b[nn] = *(const v16bf*)(Bt + (size_t)nn * ntStride + kOff);
    };
    auto mac = [&](AFrag& a, v16bf* b) {
#pragma unroll
        for (int nn = 0; nn < 4; ++nn)
            acc[nn] = __builtin_amdgcn_wmma_f32_16x16x32_bf16(
                false, a.v, false, b[nn], (short)0, acc[nn], false, false);
    };

    loadA(a0, 0);  loadB(b0, 0);
    loadA(a1, 32); loadB(b1, 32);
    for (int kt = 64; kt < K; kt += 64) {
        // keep the streamed B tiles hot in L2/WGP$: global_prefetch_b8
        __builtin_prefetch(Bt + ((((size_t)(kt >> 5)) + 16) << 9), 0, 1);
        mac(a0, b0);
        loadA(a0, kt);      loadB(b0, kt);
        mac(a1, b1);
        loadA(a1, kt + 32); loadB(b1, kt + 32);
    }
    mac(a0, b0);
    mac(a1, b1);
}

// ---------------------------------------------------------------------------
// C[64,N] = A0@B0 (+ A1@B1 if A1 != null) + bias.
// Block = 128 threads = 4 waves; wave w owns rows [16w,16w+16) x 64 cols.
// Fusing x@W and h@U into one accumulator removes the gates RMW round-trip.
// ---------------------------------------------------------------------------
__global__ __launch_bounds__(128) void gemm_dual(
    const bf16* __restrict__ A0, const bf16* __restrict__ B0, int K0,
    const bf16* __restrict__ A1, const bf16* __restrict__ B1, int K1,
    float* __restrict__ C, const float* __restrict__ bias, int N)
{
    const int lane = threadIdx.x & 31;
    const int wave = threadIdx.x >> 5;
    const int n0 = blockIdx.x << 6;
    const int m0 = wave << 4;

    // A fragment addressing (16-bit A 16x32): lane<16 -> M=lane, K 0..7/16..23;
    // lane>=16 -> M=lane-16, K 8..15/24..31  => two 16B loads at kb, kb+16.
    const int mA = m0 + (lane & 15);
    const int kb = (lane >> 4) << 3;                    // 0 or 8

    v8f acc[4] = {};

    {
        const bf16* Bt0 = B0 + (size_t)(n0 >> 4) * ((size_t)K0 * 16)
                             + (size_t)(lane << 4);
        gemm_panel(A0, mA, kb, K0, Bt0, acc);
    }
    if (A1) {   // grid-uniform branch: EXEC stays all-ones for WMMA
        const bf16* Bt1 = B1 + (size_t)(n0 >> 4) * ((size_t)K1 * 16)
                             + (size_t)(lane << 4);
        gemm_panel(A1, mA, kb, K1, Bt1, acc);
    }

    // C layout (f32 16x16): lane&15 = N col; vgpr r = row r (lanes<16) / r+8
    const int ncol  = lane & 15;
    const int mbase = m0 + ((lane >> 4) << 3);
#pragma unroll
    for (int nn = 0; nn < 4; ++nn) {
        const int n = n0 + (nn << 4) + ncol;
        const float badd = bias[n];
#pragma unroll
        for (int r = 0; r < 8; ++r)
            C[(size_t)(mbase + r) * N + n] = acc[nn][r] + badd;
    }
}

// ---------------------------------------------------------------------------

__global__ void gather_embed(const float* __restrict__ emb, const int* __restrict__ idx,
                             int t, bf16* __restrict__ x)
{
    int tid = blockIdx.x * blockDim.x + threadIdx.x;    // 64*512 threads
    int b = tid >> 9, e = tid & 511;
    x[tid] = (bf16)emb[(size_t)idx[b * TT + t] * EE + e];
}

__global__ void init_state(const float* __restrict__ cin, const float* __restrict__ hin,
                           float* __restrict__ c, bf16* __restrict__ hbf)
{
    int tid = blockIdx.x * blockDim.x + threadIdx.x;    // 64*1024 threads
    c[tid] = cin[tid];
    hbf[tid] = (bf16)hin[tid];
}

__device__ __forceinline__ float sigmoidf(float x) { return 1.0f / (1.0f + __expf(-x)); }

__global__ void lstm_cell(const float* __restrict__ gates, float* __restrict__ c,
                          bf16* __restrict__ hbf)
{
    int tid = blockIdx.x * blockDim.x + threadIdx.x;    // 64*1024 threads
    int b = tid >> 10, j = tid & 1023;
    const float* g = gates + (size_t)b * 4096;
    float gi = sigmoidf(g[j]);
    float gf = sigmoidf(g[1024 + j]);
    float gg = tanhf(g[2048 + j]);
    float go = sigmoidf(g[3072 + j]);
    float cn = gf * c[tid] + gi * gg;
    c[tid] = cn;
    hbf[tid] = (bf16)(go * tanhf(cn));
}

__global__ void softmax_row(const float* __restrict__ logits, float* __restrict__ out, int t)
{
    __shared__ float red[256];
    const int b = blockIdx.x;
    const float* row = logits + (size_t)b * VV;
    float m = -3.4e38f;
    for (int v = threadIdx.x; v < VV; v += 256) m = fmaxf(m, row[v]);
    red[threadIdx.x] = m; __syncthreads();
    for (int s = 128; s > 0; s >>= 1) {
        if (threadIdx.x < s) red[threadIdx.x] = fmaxf(red[threadIdx.x], red[threadIdx.x + s]);
        __syncthreads();
    }
    m = red[0]; __syncthreads();
    float sum = 0.0f;
    for (int v = threadIdx.x; v < VV; v += 256) sum += __expf(row[v] - m);
    red[threadIdx.x] = sum; __syncthreads();
    for (int s = 128; s > 0; s >>= 1) {
        if (threadIdx.x < s) red[threadIdx.x] += red[threadIdx.x + s];
        __syncthreads();
    }
    const float inv = 1.0f / red[0];
    float* o = out + ((size_t)b * TT + t) * VV;
    for (int v = threadIdx.x; v < VV; v += 256) o[v] = __expf(row[v] - m) * inv;
}

// ---------------------------------------------------------------------------

extern "C" void kernel_launch(void* const* d_in, const int* in_sizes, int n_in,
                              void* d_out, int out_size, void* d_ws, size_t ws_size,
                              hipStream_t stream)
{
    const int*   idx   = (const int*)  d_in[0];
    const float* h0[4] = {(const float*)d_in[1], (const float*)d_in[3],
                          (const float*)d_in[5], (const float*)d_in[7]};
    const float* c0[4] = {(const float*)d_in[2], (const float*)d_in[4],
                          (const float*)d_in[6], (const float*)d_in[8]};
    const float* emb   = (const float*)d_in[9];
    const float* W[4]  = {(const float*)d_in[10], (const float*)d_in[13],
                          (const float*)d_in[16], (const float*)d_in[19]};
    const float* U[4]  = {(const float*)d_in[11], (const float*)d_in[14],
                          (const float*)d_in[17], (const float*)d_in[20]};
    const float* bv[4] = {(const float*)d_in[12], (const float*)d_in[15],
                          (const float*)d_in[18], (const float*)d_in[21]};
    const float* Wfc = (const float*)d_in[22];
    const float* bfc = (const float*)d_in[23];
    float* out = (float*)d_out;

    char* ws = (char*)d_ws;
    size_t off = 0;
    auto alloc = [&](size_t bytes) -> void* {
        void* p = ws + off;
        off = (off + bytes + 255) & ~(size_t)255;
        return p;
    };

    bf16* Wp[4]; bf16* Up[4];
    Wp[0] = (bf16*)alloc((size_t)EE * 4 * HH * 2);
    for (int l = 1; l < 4; ++l) Wp[l] = (bf16*)alloc((size_t)HH * 4 * HH * 2);
    for (int l = 0; l < 4; ++l) Up[l] = (bf16*)alloc((size_t)HH * 4 * HH * 2);
    bf16*  Wfcp   = (bf16*) alloc((size_t)HH * VV * 2);
    float* gates  = (float*)alloc((size_t)BB * 4 * HH * 4);
    float* logits = (float*)alloc((size_t)BB * VV * 4);
    float* cs[4]; bf16* hbf[4];
    for (int l = 0; l < 4; ++l) cs[l]  = (float*)alloc((size_t)BB * HH * 4);
    for (int l = 0; l < 4; ++l) hbf[l] = (bf16*) alloc((size_t)BB * HH * 2);
    bf16* xbf = (bf16*)alloc((size_t)BB * EE * 2);

    auto pack = [&](const float* s, bf16* d, int K, int N) {
        size_t tot = (size_t)K * N;
        pack_weights<<<dim3((unsigned)((tot + 255) / 256)), dim3(256), 0, stream>>>(s, d, K, N);
    };
    pack(W[0], Wp[0], EE, 4 * HH);
    for (int l = 1; l < 4; ++l) pack(W[l], Wp[l], HH, 4 * HH);
    for (int l = 0; l < 4; ++l) pack(U[l], Up[l], HH, 4 * HH);
    pack(Wfc, Wfcp, HH, VV);

    for (int l = 0; l < 4; ++l)
        init_state<<<dim3(BB * HH / 256), dim3(256), 0, stream>>>(c0[l], h0[l], cs[l], hbf[l]);

    const int gateBlocks = (4 * HH) / 64;   // 64
    const int fcBlocks   = VV / 64;         // 500

    for (int t = 0; t < TT; ++t) {
        gather_embed<<<dim3(BB * EE / 256), dim3(256), 0, stream>>>(emb, idx, t, xbf);
        for (int l = 0; l < 4; ++l) {
            const bf16* xin = (l == 0) ? xbf : hbf[l - 1];
            const int   Kin = (l == 0) ? EE : HH;
            gemm_dual<<<dim3(gateBlocks), dim3(128), 0, stream>>>(
                xin, Wp[l], Kin, hbf[l], Up[l], HH, gates, bv[l], 4 * HH);
            lstm_cell<<<dim3(BB * HH / 256), dim3(256), 0, stream>>>(gates, cs[l], hbf[l]);
        }
        gemm_dual<<<dim3(fcBlocks), dim3(128), 0, stream>>>(
            hbf[3], Wfcp, HH, (const bf16*)nullptr, (const bf16*)nullptr, 0,
            logits, bfc, VV);
        softmax_row<<<dim3(BB), dim3(256), 0, stream>>>(logits, out, t);
    }
}
